// NGMN_53687091200569
// MI455X (gfx1250) — matrix operand order repaired and checked by reading, not compiled, not run on hardware
//
#include <hip/hip_runtime.h>
#include <hip/hip_bf16.h>
#include <math.h>

// ---------------------------------------------------------------------------
// Problem constants (from reference): B=8, N=2048, H=128
// ---------------------------------------------------------------------------
#define BB 8
#define NN 2048
#define HH 128
#define EPSF 1e-8f
#define KC 64      // K chunk for agg kernel LDS staging
#define BPAD 8     // LDS row padding (bf16 elems = 16B) to spread banks

typedef __attribute__((ext_vector_type(16))) __bf16 v16bf;
typedef __attribute__((ext_vector_type(8)))  float  v8f;
typedef __attribute__((ext_vector_type(4)))  unsigned int u32x4;

union TileU {
    v16bf v;
    u32x4 q[2];
    __bf16 h[16];
};

// A-matrix 16x32 bf16 tile loader (row-major source, K contiguous).
// Lane L holds row M=L&15; lanes 0-15 hold K {0..7,16..23},
// lanes 16-31 hold K {8..15,24..31}.  -> two b128 loads per lane.
__device__ __forceinline__ v16bf load_a_tile(const __bf16* base, int hf) {
    TileU a;
    const unsigned short* p16 = (const unsigned short*)base + (hf << 3);
    a.q[0] = *(const u32x4*)(p16);       // K = hf*8 .. hf*8+7
    a.q[1] = *(const u32x4*)(p16 + 16);  // K = 16+hf*8 .. 16+hf*8+7
    return a.v;
}

// B-matrix 32x16 bf16 tile loader; source row = output column, K contiguous.
// Lanes 0-15 hold K {0..7,16..23}, lanes 16-31 hold K {8..15,24..31}.
__device__ __forceinline__ v16bf load_b_tile(const __bf16* base, int hf) {
    TileU a;
    const unsigned short* p16 = (const unsigned short*)base + (hf << 3);
    a.q[0] = *(const u32x4*)(p16);
    a.q[1] = *(const u32x4*)(p16 + 16);
    return a.v;
}

__device__ __forceinline__ v8f wmma_bf16(v16bf a, v16bf b, v8f c) {
    return __builtin_amdgcn_wmma_f32_16x16x32_bf16(
        false, a, false, b, (short)0, c, false, false);
}

// CDNA5 async global->LDS copy (16B per lane), tracked by ASYNCcnt.
__device__ __forceinline__ void async_copy_b128(const __bf16* g, unsigned lds_off) {
    unsigned long long ga = (unsigned long long)(uintptr_t)g;
    asm volatile("global_load_async_to_lds_b128 %0, %1, off"
                 :: "v"(lds_off), "v"(ga) : "memory");
}

__device__ __forceinline__ void wait_async0() {
    asm volatile("s_wait_asynccnt 0x0" ::: "memory");
}

__device__ __forceinline__ unsigned lds_base_of(const void* p) {
    // generic pointers to LDS: low 32 bits are the LDS byte offset
    return (unsigned)(uintptr_t)p;
}

// ---------------------------------------------------------------------------
// Kernel 1: row norms -> normalized bf16 copies + transposed bf16 copies
// grid = B*N/8 blocks of 256 (8 waves), one wave32 per row
// ---------------------------------------------------------------------------
__global__ void NGMN_prep_kernel(const float* __restrict__ h1,
                                 const float* __restrict__ h2,
                                 __bf16* __restrict__ n1b, __bf16* __restrict__ n2b,
                                 __bf16* __restrict__ h1T, __bf16* __restrict__ h2T) {
    int lane = threadIdx.x & 31;
    int wave = threadIdx.x >> 5;
    int r = blockIdx.x * 8 + wave;         // 0 .. B*N-1
    int b = r >> 11;                       // / N
    int n = r & (NN - 1);
    const float* srcs[2] = {h1, h2};
    __bf16* nbs[2] = {n1b, n2b};
    __bf16* Ts[2]  = {h1T, h2T};
#pragma unroll
    for (int s = 0; s < 2; ++s) {
        const float4 x4 = *(const float4*)(srcs[s] + (size_t)r * HH + lane * 4);
        float ss = x4.x * x4.x + x4.y * x4.y + x4.z * x4.z + x4.w * x4.w;
#pragma unroll
        for (int d = 16; d >= 1; d >>= 1) ss += __shfl_xor(ss, d, 32);
        float inv = 1.0f / fmaxf(sqrtf(ss), EPSF);
        __bf16* nb = nbs[s] + (size_t)r * HH + lane * 4;
        nb[0] = (__bf16)(x4.x * inv);
        nb[1] = (__bf16)(x4.y * inv);
        nb[2] = (__bf16)(x4.z * inv);
        nb[3] = (__bf16)(x4.w * inv);
        __bf16* T = Ts[s] + (size_t)b * HH * NN + (size_t)(lane * 4) * NN + n;
        T[0 * NN] = (__bf16)x4.x;
        T[1 * NN] = (__bf16)x4.y;
        T[2 * NN] = (__bf16)x4.z;
        T[3 * NN] = (__bf16)x4.w;
    }
}

// W2 = w*w in bf16; grid 64 x 256
__global__ void NGMN_w2_kernel(const float* __restrict__ w, __bf16* __restrict__ W2b) {
    int i = blockIdx.x * 256 + threadIdx.x;   // 0..16383
    float v = w[i];
    W2b[i] = (__bf16)(v * v);
}

// ---------------------------------------------------------------------------
// Kernel 2: alpha = n1 @ n2^T per batch; write bf16 alpha and alpha^T
// grid = (N/64, N/64, B), block 128 (4 waves); wave -> 16x64 strip.
// B panel (64 rows x 128 K = 16KB, contiguous in n2b) staged once in LDS
// via async copy, shared by all 4 waves.
// ---------------------------------------------------------------------------
__global__ void NGMN_alpha_kernel(const __bf16* __restrict__ n1b,
                                  const __bf16* __restrict__ n2b,
                                  __bf16* __restrict__ alphaB,
                                  __bf16* __restrict__ alphaT) {
    __shared__ __bf16 shB[64 * (HH + BPAD)];       // 17,408 B
    int lane = threadIdx.x & 31;
    int wave = threadIdx.x >> 5;
    int hf = lane >> 4, l15 = lane & 15;
    int b = blockIdx.z;
    int i0 = blockIdx.x * 64 + wave * 16;
    int j0 = blockIdx.y * 64;
    int tid = threadIdx.x;

    // async-stage B panel: 64 rows x 128 elems; 1024 b128 segs / 128 threads = 8
    {
        const __bf16* gB = n2b + ((size_t)b * NN + j0) * HH;
        unsigned lbase = lds_base_of(&shB[0]);
#pragma unroll
        for (int s = 0; s < 8; ++s) {
            int seg = tid + s * 128;               // 0..1023
            int row = seg >> 4;                    // 0..63
            int k8  = seg & 15;                    // 0..15 (8 elems each)
            async_copy_b128(gB + (size_t)row * HH + k8 * 8,
                            lbase + (unsigned)(row * (HH + BPAD) + k8 * 8) * 2);
        }
    }
    wait_async0();
    __syncthreads();

    const __bf16* Abase = n1b + ((size_t)b * NN + i0 + l15) * HH;
    v8f acc[4] = {};
#pragma unroll
    for (int kk = 0; kk < HH; kk += 32) {
        v16bf a = load_a_tile(Abase + kk, hf);
        v16bf bt[4];
#pragma unroll
        for (int t = 0; t < 4; ++t)
            bt[t] = load_b_tile(&shB[(t * 16 + l15) * (HH + BPAD) + kk], hf);
#pragma unroll
        for (int t = 0; t < 4; ++t)
            acc[t] = wmma_bf16(a, bt[t], acc[t]);
    }
#pragma unroll
    for (int t = 0; t < 4; ++t) {
#pragma unroll
        for (int v = 0; v < 8; ++v) {
            int m = i0 + v + hf * 8;            // C layout: VGPR v, M = v + (lane/16)*8
            int j = j0 + t * 16 + l15;          // N = lane&15
            __bf16 val = (__bf16)acc[t][v];
            alphaB[((size_t)b * NN + m) * NN + j] = val;
            alphaT[((size_t)b * NN + j) * NN + m] = val;
        }
    }
}

// ---------------------------------------------------------------------------
// Kernel 3: out[M,128] = A[M,2048] @ BT[128,2048]^T  (per batch)
// grid = (M/64, 1, B), block 128 (4 waves); wave -> 16 rows x 128 cols.
// B chunk (128 cols x 64 K = 16KB) double-buffered in LDS with async copies;
// prefetch of chunk c+1 overlaps the 16 WMMAs of chunk c.  Per k-step the 8
// B tiles are hoisted into distinct registers so ds_loads clause together
// and WMMAs retire back-to-back instead of waiting dscnt==0 each.
// ---------------------------------------------------------------------------
__global__ void NGMN_agg_kernel(const __bf16* __restrict__ A,
                                const __bf16* __restrict__ BT,
                                float* __restrict__ out) {
    __shared__ __bf16 shB[2][HH * (KC + BPAD)];    // 2 x 18,432 B
    int lane = threadIdx.x & 31;
    int wave = threadIdx.x >> 5;
    int hf = lane >> 4, l15 = lane & 15;
    int b = blockIdx.z;
    int i0 = blockIdx.x * 64 + wave * 16;
    int tid = threadIdx.x;

    const __bf16* Abase  = A + ((size_t)b * NN + i0 + l15) * NN;
    const __bf16* Bbatch = BT + (size_t)b * HH * NN;

    // stage one K-chunk: 128 rows x KC elems; 1024 b128 segs / 128 threads = 8
    auto copy_chunk = [&](int buf, int kk) {
        unsigned lbase = lds_base_of(&shB[buf][0]);
#pragma unroll
        for (int s = 0; s < 8; ++s) {
            int seg = tid + s * 128;               // 0..1023
            int row = seg >> 3;                    // 0..127 (output column)
            int k8  = seg & 7;                     // 0..7
            async_copy_b128(Bbatch + (size_t)row * NN + kk + k8 * 8,
                            lbase + (unsigned)(row * (KC + BPAD) + k8 * 8) * 2);
        }
    };

    v8f acc[8] = {};
    copy_chunk(0, 0);
    for (int c = 0; c < NN / KC; ++c) {
        wait_async0();
        __syncthreads();                           // chunk c resident for all waves
        if (c + 1 < NN / KC) copy_chunk((c + 1) & 1, (c + 1) * KC);
        const __bf16* shb = &shB[c & 1][0];

        // A tiles first (global, fly while LDS is read)
        v16bf a0 = load_a_tile(Abase + c * KC, hf);
        v16bf a1 = load_a_tile(Abase + c * KC + 32, hf);

        v16bf bt[8];
#pragma unroll
        for (int t = 0; t < 8; ++t)
            bt[t] = load_b_tile(shb + (t * 16 + l15) * (KC + BPAD), hf);
#pragma unroll
        for (int t = 0; t < 8; ++t)
            acc[t] = wmma_bf16(a0, bt[t], acc[t]);
#pragma unroll
        for (int t = 0; t < 8; ++t)
            bt[t] = load_b_tile(shb + (t * 16 + l15) * (KC + BPAD) + 32, hf);
#pragma unroll
        for (int t = 0; t < 8; ++t)
            acc[t] = wmma_bf16(a1, bt[t], acc[t]);
    }
#pragma unroll
    for (int t = 0; t < 8; ++t) {
#pragma unroll
        for (int v = 0; v < 8; ++v) {
            int m = i0 + v + hf * 8;
            out[((size_t)b * NN + m) * HH + t * 16 + l15] = acc[t][v];
        }
    }
}

// ---------------------------------------------------------------------------
// Kernel 4: _fm epilogue.
// num = (x*y)@W2^T, dx=sqrt((x*x)@W2^T), dy=sqrt((y*y)@W2^T)
// out = num / (max(dx,eps)*max(dy,eps)).  Rows flat over B*N.
// grid = B*N/64, block 128; W2 (32KB) staged once in LDS, reused by all
// K iterations and all three accumulator streams.
// ---------------------------------------------------------------------------
__global__ void NGMN_fm_kernel(const float* __restrict__ x,
                               const float* __restrict__ y,
                               const __bf16* __restrict__ W2b,
                               float* __restrict__ out) {
    __shared__ __bf16 shW[HH * (HH + BPAD)];       // 34,816 B
    int lane = threadIdx.x & 31;
    int wave = threadIdx.x >> 5;
    int hf = lane >> 4, l15 = lane & 15;
    int tid = threadIdx.x;
    size_t r0 = (size_t)blockIdx.x * 64 + wave * 16;
    size_t row = r0 + l15;                 // A layout: lane holds row lane&15

    // async-stage W2: 128 rows x 128 elems; 2048 b128 segs / 128 threads = 16
    {
        unsigned lbase = lds_base_of(&shW[0]);
#pragma unroll
        for (int s = 0; s < 16; ++s) {
            int seg = tid + s * 128;               // 0..2047
            int rw = seg >> 4;                     // 0..127
            int k8 = seg & 15;                     // 0..15
            async_copy_b128(W2b + (size_t)rw * HH + k8 * 8,
                            lbase + (unsigned)(rw * (HH + BPAD) + k8 * 8) * 2);
        }
    }

    const float* xr = x + row * HH;
    const float* yr = y + row * HH;

    wait_async0();
    __syncthreads();

    v8f num[8] = {}, dxx[8] = {}, dyy[8] = {};
#pragma unroll
    for (int kk = 0; kk < HH; kk += 32) {
        TileU ap, aq, ar;
        // A K-interleave: group g covers K = kk + g*16 + hf*8 .. +7 (8 contiguous)
#pragma unroll
        for (int g = 0; g < 2; ++g) {
            int kbase = kk + g * 16 + (hf << 3);
            float xv[8], yv[8];
            *(float4*)&xv[0] = *(const float4*)(xr + kbase);
            *(float4*)&xv[4] = *(const float4*)(xr + kbase + 4);
            *(float4*)&yv[0] = *(const float4*)(yr + kbase);
            *(float4*)&yv[4] = *(const float4*)(yr + kbase + 4);
#pragma unroll
            for (int e = 0; e < 8; ++e) {
                ap.h[g * 8 + e] = (__bf16)(xv[e] * yv[e]);
                aq.h[g * 8 + e] = (__bf16)(xv[e] * xv[e]);
                ar.h[g * 8 + e] = (__bf16)(yv[e] * yv[e]);
            }
        }
#pragma unroll
        for (int t = 0; t < 8; ++t) {
            // B[l,k] = W2[k,l] -> lane (out-channel) reads its W2 row, K-contiguous
            v16bf bw = load_b_tile(&shW[(t * 16 + l15) * (HH + BPAD) + kk], hf);
            num[t] = wmma_bf16(ap.v, bw, num[t]);
            dxx[t] = wmma_bf16(aq.v, bw, dxx[t]);
            dyy[t] = wmma_bf16(ar.v, bw, dyy[t]);
        }
    }
#pragma unroll
    for (int t = 0; t < 8; ++t) {
#pragma unroll
        for (int v = 0; v < 8; ++v) {
            size_t m = r0 + v + hf * 8;
            float dx = fmaxf(sqrtf(fmaxf(dxx[t][v], 0.0f)), EPSF);
            float dy = fmaxf(sqrtf(fmaxf(dyy[t][v], 0.0f)), EPSF);
            out[m * HH + t * 16 + l15] = num[t][v] / (dx * dy);
        }
    }
}

// ---------------------------------------------------------------------------
// Launch
// ---------------------------------------------------------------------------
extern "C" void kernel_launch(void* const* d_in, const int* in_sizes, int n_in,
                              void* d_out, int out_size, void* d_ws, size_t ws_size,
                              hipStream_t stream) {
    (void)in_sizes; (void)n_in; (void)out_size; (void)ws_size;
    const float* h1  = (const float*)d_in[0];
    const float* h2  = (const float*)d_in[1];
    const float* w_m = (const float*)d_in[2];
    float* out = (float*)d_out;

    const size_t BNH = (size_t)BB * NN * HH;       // 2,097,152
    const size_t BNN = (size_t)BB * NN * NN;       // 33,554,432

    char* ws = (char*)d_ws;
    __bf16* n1b    = (__bf16*)ws;                 ws += BNH * 2;      // 4 MiB
    __bf16* n2b    = (__bf16*)ws;                 ws += BNH * 2;      // 4 MiB
    __bf16* h1T    = (__bf16*)ws;                 ws += BNH * 2;      // 4 MiB
    __bf16* h2T    = (__bf16*)ws;                 ws += BNH * 2;      // 4 MiB
    __bf16* W2b    = (__bf16*)ws;                 ws += (size_t)HH * HH * 2 + 256;
    __bf16* alphaB = (__bf16*)ws;                 ws += BNN * 2;      // 64 MiB
    __bf16* alphaT = (__bf16*)ws;                 ws += BNN * 2;      // 64 MiB
    float*  hg2    = (float*)ws;                  ws += BNH * 4;      // 8 MiB
    float*  hg1    = (float*)ws;                  ws += BNH * 4;      // 8 MiB

    // 1. norms + bf16 / transposed copies; W2 = w*w
    NGMN_prep_kernel<<<dim3((BB * NN) / 8), 256, 0, stream>>>(h1, h2, n1b, n2b, h1T, h2T);
    NGMN_w2_kernel<<<dim3((HH * HH) / 256), 256, 0, stream>>>(w_m, W2b);

    // 2. alpha = n1 @ n2^T  (and transposed copy)
    NGMN_alpha_kernel<<<dim3(NN / 64, NN / 64, BB), 128, 0, stream>>>(n1b, n2b, alphaB, alphaT);

    // 3. h_g2 = alpha @ h2 ; h_g1 = alpha^T @ h1
    NGMN_agg_kernel<<<dim3(NN / 64, 1, BB), 128, 0, stream>>>(alphaB, h2T, hg2);
    NGMN_agg_kernel<<<dim3(NN / 64, 1, BB), 128, 0, stream>>>(alphaT, h1T, hg1);

    // 4. out1 = fm(h1, h_g2) ; out2 = fm(h2, h_g1)
    NGMN_fm_kernel<<<dim3((BB * NN) / 64), 128, 0, stream>>>(h1, hg2, W2b, out);
    NGMN_fm_kernel<<<dim3((BB * NN) / 64), 128, 0, stream>>>(h2, hg1, W2b, out + BNH);
}
